// RecBlock_36481452212422
// MI455X (gfx1250) — matrix operand order repaired and checked by reading, not compiled, not run on hardware
//
#include <hip/hip_runtime.h>
#include <cstdint>

// ---------------- problem constants ----------------
#define NWG    16
#define NTHR   512          // 16 waves of 32
#define NWAVE  16
#define BATCH  256
#define SEQ    2048
#define NIN    128
#define HDIM   256
#define NOUT   128
#define KTOT   384          // NIN + HDIM
#define BN_EPS 1e-5f

// LDS pitches (in f16 elements); multiples of 8 so rows stay 16B-aligned,
// padded (+8) to spread LDS banks. XH row = [x:256B][h:512B][pad:16B] = 784B.
#define XH_PITCH  392
#define WIH_PITCH 392
#define WHO_PITCH 264

// LDS layout, half-element offsets
#define OFF_XH     0
#define OFF_WIH_HI (BATCH * XH_PITCH)                 // 100352
#define OFF_WIH_LO (OFF_WIH_HI + 16 * WIH_PITCH)
#define OFF_WHO_HI (OFF_WIH_LO + 16 * WIH_PITCH)
#define OFF_WHO_LO (OFF_WHO_HI + 16 * WHO_PITCH)
#define HALVES_TOT (OFF_WHO_LO + 16 * WHO_PITCH)      // 121344 halves
// float scratch after the f16 region
#define F_RED1   0
#define F_RED2   256
#define F_SCALE  512
#define F_SHIFT  528
#define F_BHO    544
#define F_GAMMA  560
#define F_BETA   576
#define FLOATS_TOT 592
#define SMEM_BYTES (HALVES_TOT * 2 + FLOATS_TOT * 4)  // 245056 B < 320KB/WGP

typedef __attribute__((ext_vector_type(16))) _Float16 v16h;
typedef __attribute__((ext_vector_type(8)))  float    v8f;
typedef __attribute__((ext_vector_type(4)))  unsigned uint4v;
typedef __attribute__((ext_vector_type(4)))  unsigned u32x4;
typedef __attribute__((ext_vector_type(8)))  int      i32x8;
typedef __attribute__((ext_vector_type(4)))  int      i32x4;

union Frag16 { uint4v q[2]; v16h h; };

// ---------------- fragment loaders (contiguous K = 16h + s, same order in A and B) ----------
// A (16x32 f16): lane L -> m = rb + (L&15); lanes<16 carry K 0..15,
// lanes>=16 carry K 16..31; 16 contiguous K halves -> 2x ds_read_b128.
__device__ __forceinline__ v16h load_a(const _Float16* smem, int rb, int kbase, int lane) {
  const int m  = rb + (lane & 15);
  const int kb = (lane >> 4) << 4;                   // 0 or 16
  const _Float16* p = smem + (size_t)m * XH_PITCH + kbase + kb;
  Frag16 f;
  f.q[0] = *(const uint4v*)(p);
  f.q[1] = *(const uint4v*)(p + 8);
  return f.h;
}
// B (32x16 f16, K x N): lane L -> n = L&15; lanes<16 K=0..15, lanes>=16 K=16..31,
// 16 contiguous K halves -> 2x ds_read_b128. W stored row-major [feature][k].
__device__ __forceinline__ v16h load_b(const _Float16* w, int pitch, int kbase, int lane) {
  const _Float16* p = w + (size_t)(lane & 15) * pitch + kbase + ((lane >> 4) << 4);
  Frag16 f;
  f.q[0] = *(const uint4v*)(p);
  f.q[1] = *(const uint4v*)(p + 8);
  return f.h;
}

__device__ __forceinline__ float gelu_exact(float x) {
  return 0.5f * x * (1.0f + erff(x * 0.7071067811865476f));
}

// ---------------- grid barrier (16 persistent WGs) ----------------
__device__ __forceinline__ void grid_arrive(unsigned* cnt, unsigned* gen, unsigned epoch) {
  unsigned a = __hip_atomic_fetch_add(cnt, 1u, __ATOMIC_ACQ_REL, __HIP_MEMORY_SCOPE_AGENT);
  if (a == epoch * NWG - 1u)
    __hip_atomic_store(gen, epoch, __ATOMIC_RELEASE, __HIP_MEMORY_SCOPE_AGENT);
}
__device__ __forceinline__ void grid_wait(unsigned* gen, unsigned epoch) {
  while (__hip_atomic_load(gen, __ATOMIC_ACQUIRE, __HIP_MEMORY_SCOPE_AGENT) < epoch)
    __builtin_amdgcn_s_sleep(1);
}

// ---------------- TDM: DMA h (256x256 f16) global -> LDS h-columns ----------
// D# per CDNA5 ISA ch.8: 2-D tensor, data_size=2B, LDS padding of 68 DWORDs
// after every 128 DWORDs (= one 512B h-row) lands rows at the 784B XH pitch
// while *skipping* (preserving) the interleaved x-columns and row tail.
__device__ __forceinline__ void tdm_load_h(const _Float16* smem, const _Float16* hsrc) {
  const uint32_t ldsa = (uint32_t)(uintptr_t)(smem + NIN);   // flat addr[31:0] == LDS offset
  const uint64_t ga   = (uint64_t)(uintptr_t)hsrc;
  u32x4 g0;
  g0[0] = 1u;                                                // count=1, user descriptor
  g0[1] = ldsa;                                              // lds_addr (bytes)
  g0[2] = (uint32_t)ga;                                      // global_addr[31:0]
  g0[3] = (uint32_t)((ga >> 32) & 0x01FFFFFFu) | (2u << 30); // global_addr[56:32] | type=2
  i32x8 g1;
  g1[0] = (int)((1u << 16) |        // data_size = 2B
                (1u << 20) |        // pad_enable
                (6u << 22) |        // pad_interval: 128 DWORDs
                (67u << 25));       // pad_amount: 68 DWORDs (272B)
  g1[1] = (int)(256u << 16);        // tensor_dim0 = 256
  g1[2] = (int)(256u << 16);        // tensor_dim1 = 256
  g1[3] = (int)(256u << 16);        // tile_dim0   = 256
  g1[4] = 256;                      // tile_dim1   = 256 (tile_dim2 = 0)
  g1[5] = 256;                      // tensor_dim0_stride = 256
  g1[6] = 0;
  g1[7] = 0;
  const i32x4 z4 = {0, 0, 0, 0};            // groups 2/3: zero (2-D tensor)
  const i32x8 z8 = {0, 0, 0, 0, 0, 0, 0, 0};
  __builtin_amdgcn_tensor_load_to_lds(g0, g1, z4, z4, z8, 0);
}

// ---------------- x staging (needs f32 -> f16 convert, stays vector) -------
__device__ __forceinline__ void stage_x_lds(_Float16* smem, const float* X, int t, int tid) {
  for (int i = tid; i < (BATCH * NIN) / 4; i += NTHR) {
    const int b = i >> 5;
    const int c = (i & 31) << 2;
    const float4 v = *(const float4*)(X + ((size_t)b * SEQ + t) * NIN + c);
    union { _Float16 h[4]; uint64_t u; } p;
    p.h[0] = (_Float16)v.x; p.h[1] = (_Float16)v.y;
    p.h[2] = (_Float16)v.z; p.h[3] = (_Float16)v.w;
    *(uint64_t*)(smem + (size_t)b * XH_PITCH + c) = p.u;
  }
}

// ---------------- o = gelu(h @ W_ho^T + b_ho) for one step ----------------
__device__ __forceinline__ void o_step(const _Float16* smem, const float* fsm,
                                       float* __restrict__ out, int tprev,
                                       int rb, int lane) {
  const int l16 = lane & 15;
  const int lh  = lane >> 4;
  v8f co = {};
  #pragma unroll
  for (int kk = 0; kk < 8; ++kk) {
    const int kb = kk * 32;
    v16h bh = load_b(smem + OFF_WHO_HI, WHO_PITCH, kb, lane);
    v16h bl = load_b(smem + OFF_WHO_LO, WHO_PITCH, kb, lane);
    v16h a  = load_a(smem, rb, NIN + kb, lane);      // h part of [x|h]
    co = __builtin_amdgcn_wmma_f32_16x16x32_f16(false, a, false, bh, (short)0, co, false, false);
    co = __builtin_amdgcn_wmma_f32_16x16x32_f16(false, a, false, bl, (short)0, co, false, false);
  }
  const float bias = fsm[F_BHO + l16];
  const int g8 = blockIdx.x * 8;
  if (l16 < 8) {
    #pragma unroll
    for (int j = 0; j < 8; ++j) {
      const int b = rb + lh * 8 + j;
      out[((size_t)b * SEQ + tprev) * NOUT + g8 + l16] = gelu_exact(co[j] + bias);
    }
  }
}

// ---------------- persistent recurrent kernel ----------------
extern "C" __global__ void __launch_bounds__(NTHR, 1)
rec_kernel(const float* __restrict__ X, const float* __restrict__ W_ih,
           const float* __restrict__ W_ho, const float* __restrict__ b_ho,
           const float* __restrict__ gamma, const float* __restrict__ beta,
           float* __restrict__ out, _Float16* __restrict__ hglob,
           unsigned* __restrict__ sync) {
  extern __shared__ _Float16 smem[];
  float* fsm = (float*)(smem + HALVES_TOT);

  const int tid  = threadIdx.x;
  const int g    = blockIdx.x;
  const int lane = tid & 31;
  const int wv   = tid >> 5;
  const int l16  = lane & 15;
  const int lh   = lane >> 4;
  const int rb   = wv * 16;          // this wave's 16-row tile
  const int g16  = g * 16;
  const int g8   = g * 8;

  unsigned* cnt = sync;
  unsigned* gen = sync + 1;
  unsigned epoch = 0;

  _Float16* h0 = hglob;
  _Float16* h1 = hglob + BATCH * HDIM;

  // ---- one-time: split-f16 weights into LDS ----
  for (int i = tid; i < 16 * KTOT; i += NTHR) {
    const int f = i / KTOT, k = i - f * KTOT;
    const float w = W_ih[(size_t)(g16 + f) * KTOT + k];
    const _Float16 hi = (_Float16)w;
    smem[OFF_WIH_HI + f * WIH_PITCH + k] = hi;
    smem[OFF_WIH_LO + f * WIH_PITCH + k] = (_Float16)(w - (float)hi);
  }
  for (int i = tid; i < 16 * HDIM; i += NTHR) {
    const int f = i >> 8, k = i & 255;
    const float w = (f < 8) ? W_ho[(size_t)(g8 + f) * HDIM + k] : 0.0f;
    const _Float16 hi = (_Float16)w;
    smem[OFF_WHO_HI + f * WHO_PITCH + k] = hi;
    smem[OFF_WHO_LO + f * WHO_PITCH + k] = (_Float16)(w - (float)hi);
  }
  if (tid < 16) {
    fsm[F_BHO   + tid] = (tid < 8) ? b_ho[g8 + tid] : 0.0f;
    fsm[F_GAMMA + tid] = gamma[g16 + tid];
    fsm[F_BETA  + tid] = beta[g16 + tid];
  }
  // NOTE: b_ih is intentionally unused — BatchNorm subtracts the batch mean,
  // which absorbs any per-feature constant bias exactly.

  // ---- zero h_{-1} (grid-wide) then grid barrier ----
  {
    unsigned* hz = (unsigned*)h0;
    for (int i = g * NTHR + tid; i < (BATCH * HDIM) / 2; i += NWG * NTHR) hz[i] = 0u;
  }
  __threadfence();
  __syncthreads();
  ++epoch;
  if (tid == 0) { grid_arrive(cnt, gen, epoch); grid_wait(gen, epoch); }
  __syncthreads();

  stage_x_lds(smem, X, 0, tid);

  int buf = 0;
  for (int t = 0; t < SEQ; ++t) {
    const _Float16* hcur = buf ? h1 : h0;
    _Float16* hnext      = buf ? h0 : h1;

    // ---- h_{t-1} -> LDS via Tensor Data Mover (wave 0 issues; TENSORcnt) ----
    if (wv == 0) {
      tdm_load_h(smem, hcur);
      __builtin_amdgcn_s_wait_tensorcnt(0);
    }
    __syncthreads();

    // ---- z = [x_t | h_{t-1}] @ W_ih^T  (split-f16 weights, f32 acc) ----
    v8f cz = {};
    #pragma unroll
    for (int kk = 0; kk < 12; ++kk) {
      const int kb = kk * 32;
      v16h bh = load_b(smem + OFF_WIH_HI, WIH_PITCH, kb, lane);
      v16h bl = load_b(smem + OFF_WIH_LO, WIH_PITCH, kb, lane);
      v16h a  = load_a(smem, rb, kb, lane);
      cz = __builtin_amdgcn_wmma_f32_16x16x32_f16(false, a, false, bh, (short)0, cz, false, false);
      cz = __builtin_amdgcn_wmma_f32_16x16x32_f16(false, a, false, bl, (short)0, cz, false, false);
    }

    // ---- BatchNorm stats over batch (local: features are WG-private) ----
    float s1 = 0.0f, s2 = 0.0f;
    #pragma unroll
    for (int j = 0; j < 8; ++j) { const float v = cz[j]; s1 += v; s2 += v * v; }
    s1 += __shfl_xor(s1, 16, 32);
    s2 += __shfl_xor(s2, 16, 32);
    if (lh == 0) { fsm[F_RED1 + wv * 16 + l16] = s1; fsm[F_RED2 + wv * 16 + l16] = s2; }
    __syncthreads();
    if (tid < 16) {
      float m = 0.0f, q = 0.0f;
      #pragma unroll
      for (int w = 0; w < NWAVE; ++w) { m += fsm[F_RED1 + w * 16 + tid]; q += fsm[F_RED2 + w * 16 + tid]; }
      m *= (1.0f / 256.0f);
      q  = q * (1.0f / 256.0f) - m * m;                // biased variance
      const float sc = fsm[F_GAMMA + tid] * rsqrtf(q + BN_EPS);
      fsm[F_SCALE + tid] = sc;
      fsm[F_SHIFT + tid] = fsm[F_BETA + tid] - m * sc;
    }
    __syncthreads();

    // ---- normalize + exact GELU -> h_t, publish to other WGs via L2 ----
    {
      const float sc = fsm[F_SCALE + l16];
      const float sh = fsm[F_SHIFT + l16];
      #pragma unroll
      for (int j = 0; j < 8; ++j) {
        const float hn = gelu_exact(cz[j] * sc + sh);
        hnext[(size_t)(rb + lh * 8 + j) * HDIM + g16 + l16] = (_Float16)hn;
      }
    }

    // ---- grid barrier: arrive, then hide o_{t-1} + x_{t+1} staging in wait ----
    __threadfence();
    __syncthreads();
    ++epoch;
    if (tid == 0) grid_arrive(cnt, gen, epoch);

    if (t + 2 < SEQ && tid < BATCH)
      __builtin_prefetch(X + ((size_t)tid * SEQ + (t + 2)) * NIN, 0, 1);
    if (t + 1 < SEQ) stage_x_lds(smem, X, t + 1, tid);     // x-cols only; o reads h-cols
    if (t >= 1)      o_step(smem, fsm, out, t - 1, rb, lane);

    if (tid == 0) grid_wait(gen, epoch);
    __syncthreads();
    buf ^= 1;
  }

  // ---- epilogue: o_{S-1} from the final h ----
  if (wv == 0) {
    tdm_load_h(smem, buf ? h1 : h0);
    __builtin_amdgcn_s_wait_tensorcnt(0);
  }
  __syncthreads();
  o_step(smem, fsm, out, SEQ - 1, rb, lane);
}

// ---------------- host launch ----------------
extern "C" void kernel_launch(void* const* d_in, const int* in_sizes, int n_in,
                              void* d_out, int out_size, void* d_ws, size_t ws_size,
                              hipStream_t stream) {
  const float* X     = (const float*)d_in[0];
  const float* W_ih  = (const float*)d_in[1];
  // d_in[2] = b_ih : mathematically absorbed by BatchNorm mean subtraction
  const float* W_ho  = (const float*)d_in[3];
  const float* b_ho  = (const float*)d_in[4];
  const float* gamma = (const float*)d_in[5];
  const float* beta  = (const float*)d_in[6];
  float* out = (float*)d_out;

  unsigned*  sync  = (unsigned*)d_ws;
  _Float16*  hglob = (_Float16*)((char*)d_ws + 256);   // 2 x (256*256) f16 buffers

  (void)hipMemsetAsync(d_ws, 0, 256, stream);          // reset barrier state each call
  (void)hipFuncSetAttribute(reinterpret_cast<const void*>(rec_kernel),
                            hipFuncAttributeMaxDynamicSharedMemorySize, SMEM_BYTES);
  rec_kernel<<<dim3(NWG), dim3(NTHR), SMEM_BYTES, stream>>>(
      X, W_ih, W_ho, b_ho, gamma, beta, out, hglob, sync);
}
